// Weights_91259465105912
// MI455X (gfx1250) — matrix operand order repaired and checked by reading, not compiled, not run on hardware
//
#include <hip/hip_runtime.h>
#include <cstdint>
#include <cstddef>

// ---- problem sizes ----
constexpr int T_  = 1024;
constexpr int B_  = 4;
constexpr int D_  = 1024;
constexpr int NH_ = 8;
constexpr int DH_ = 128;
constexpr int MEM_ = 1024;
constexpr int TM_ = T_ + MEM_;    // 2048
constexpr int SROW = 1056;        // band width in LDS (66 tiles of 16)

typedef __attribute__((ext_vector_type(16))) __bf16 bf16x16;
typedef __attribute__((ext_vector_type(8)))  float  f32x8;

// float -> bf16 bits, round-to-nearest-even
__device__ __forceinline__ uint16_t f2bf(float f) {
    uint32_t u = __float_as_uint(f);
    u += 0x7FFFu + ((u >> 16) & 1u);
    return (uint16_t)(u >> 16);
}

// build a 16-element bf16 fragment from two 16-byte chunks
__device__ __forceinline__ bf16x16 frag2(const uint16_t* p0, const uint16_t* p1) {
    union { uint32_t u[8]; bf16x16 v; } r;
    uint4 a = *reinterpret_cast<const uint4*>(p0);
    uint4 b = *reinterpret_cast<const uint4*>(p1);
    r.u[0]=a.x; r.u[1]=a.y; r.u[2]=a.z; r.u[3]=a.w;
    r.u[4]=b.x; r.u[5]=b.y; r.u[6]=b.z; r.u[7]=b.w;
    return r.v;
}
// A fragment (16xK, this lane's row pointer), K chunk at k0
__device__ __forceinline__ bf16x16 loadA(const uint16_t* row, int k0, int kb) {
    return frag2(row + k0 + kb, row + k0 + kb + 16);
}
// B fragment (Kx16, this lane's column-row pointer), K chunk at k0
__device__ __forceinline__ bf16x16 loadB(const uint16_t* row, int k0, int ks) {
    return frag2(row + k0 + ks / 2, row + k0 + ks / 2 + 16);
}
__device__ __forceinline__ f32x8 wmma_bf16(bf16x16 a, bf16x16 b, f32x8 c) {
    return __builtin_amdgcn_wmma_f32_16x16x32_bf16(false, a, false, b, (short)0, c, false, false);
}

// ---------------- stage 0: layernorm -> bf16 ----------------
__global__ __launch_bounds__(256) void ln_kernel(
    const float* __restrict__ h, const float* __restrict__ m,
    const float* __restrict__ gamma, const float* __restrict__ beta,
    uint16_t* __restrict__ out)
{
    const int row = blockIdx.x;     // t*B + b, t in [0,TM)
    const float* src = (row < MEM_ * B_) ? (m + (size_t)row * D_)
                                         : (h + (size_t)(row - MEM_ * B_) * D_);
    const int tid = threadIdx.x, lane = tid & 31, w = tid >> 5;
    float x[4]; float s = 0.f, ss = 0.f;
#pragma unroll
    for (int u = 0; u < 4; ++u) { int d = tid + 256 * u; float v = src[d]; x[u] = v; s += v; ss += v * v; }
#pragma unroll
    for (int off = 16; off; off >>= 1) { s += __shfl_xor(s, off, 32); ss += __shfl_xor(ss, off, 32); }
    __shared__ float r1[8], r2[8];
    if (lane == 0) { r1[w] = s; r2[w] = ss; }
    __syncthreads();
    s = 0.f; ss = 0.f;
#pragma unroll
    for (int u = 0; u < 8; ++u) { s += r1[u]; ss += r2[u]; }
    float mu = s * (1.0f / D_);
    float var = ss * (1.0f / D_) - mu * mu;
    float rstd = rsqrtf(var + 1e-5f);
#pragma unroll
    for (int u = 0; u < 4; ++u) {
        int d = tid + 256 * u;
        float y = (x[u] - mu) * rstd * gamma[d] + beta[d];
        out[(size_t)row * D_ + d] = f2bf(y);
    }
}

// ---------------- positional encoding -> bf16 ----------------
__global__ __launch_bounds__(256) void pe_kernel(uint16_t* __restrict__ pe) {
    const int t = blockIdx.x;                    // 0..TM-1
    const float pos = (float)(TM_ - 1 - t);
    for (int d = threadIdx.x; d < D_; d += 256) {
        int x = (d < D_ / 2) ? d : d - D_ / 2;
        // inv_freq = 2*pi / 1024^(2x/1024) = 2*pi * 2^(-20x/1024)
        float f = 6.2831853071795864f * exp2f(-(float)x * (20.0f / 1024.0f));
        float ang = pos * f;
        float v = (d < D_ / 2) ? __sinf(ang) : __cosf(ang);
        pe[(size_t)t * D_ + d] = f2bf(v);
    }
}

// ---------------- fp32 -> bf16 weight convert ----------------
__global__ __launch_bounds__(256) void cvt_kernel(const float* __restrict__ src,
                                                  uint16_t* __restrict__ dst, int n) {
    int i = blockIdx.x * 256 + threadIdx.x;
    if (i < n) dst[i] = f2bf(src[i]);
}

// ---------------- reset cumsum ----------------
__global__ void cum_kernel(const int* __restrict__ reset, const int* __restrict__ old_reset,
                           int* __restrict__ cum) {
    int b = threadIdx.x;
    if (b >= B_) return;
    int c = 0;
    for (int t = 0; t < TM_; ++t) {
        int r = (t < MEM_) ? old_reset[t * B_ + b] : reset[(t - MEM_) * B_ + b];
        c += (r != 0);
        cum[t * B_ + b] = c;
    }
}

// ---------------- bf16 WMMA 2x2-tile GEMM core (wave: 32M x 32N, K=1024) ----------------
// Each fragment load feeds two WMMAs: 16 FLOP/B from cache vs 8 for a 1x1 tile.
__device__ __forceinline__ void gemm22(const uint16_t* __restrict__ A,
                                       const uint16_t* __restrict__ W,
                                       int m0, int n0, int kb, int ks, int sl,
                                       f32x8 acc[4]) {
    const uint16_t* a0 = A + (size_t)(m0 + sl) * D_;
    const uint16_t* a1 = A + (size_t)(m0 + 16 + sl) * D_;
    const uint16_t* b0 = W + (size_t)(n0 + sl) * D_;
    const uint16_t* b1 = W + (size_t)(n0 + 16 + sl) * D_;
#pragma unroll 2
    for (int k0 = 0; k0 < D_; k0 += 32) {
        bf16x16 va0 = loadA(a0, k0, kb);
        bf16x16 va1 = loadA(a1, k0, kb);
        bf16x16 vb0 = loadB(b0, k0, ks);
        bf16x16 vb1 = loadB(b1, k0, ks);
        acc[0] = wmma_bf16(va0, vb0, acc[0]);
        acc[1] = wmma_bf16(va0, vb1, acc[1]);
        acc[2] = wmma_bf16(va1, vb0, acc[2]);
        acc[3] = wmma_bf16(va1, vb1, acc[3]);
    }
}
#define ACC4_INIT { {0.f,0.f,0.f,0.f,0.f,0.f,0.f,0.f}, {0.f,0.f,0.f,0.f,0.f,0.f,0.f,0.f}, \
                    {0.f,0.f,0.f,0.f,0.f,0.f,0.f,0.f}, {0.f,0.f,0.f,0.f,0.f,0.f,0.f,0.f} }

// K projection: cat(8192x1024) @ k_w^T -> kbuf[b][nh][t][dh] (bf16). Grid (M/64, N/128).
__global__ __launch_bounds__(256) void gemm_k_kernel(
    const uint16_t* __restrict__ cat, const uint16_t* __restrict__ kw,
    uint16_t* __restrict__ kout)
{
    const int lane = threadIdx.x & 31, w = threadIdx.x >> 5;
    const int kb = (lane >> 4) * 8, ks = (lane >> 4) * 16, sl = lane & 15;
    const int hi = (lane >> 4) << 3;
    const int m0 = blockIdx.x * 64 + (w >> 2) * 32;
    const int n0 = blockIdx.y * 128 + (w & 3) * 32;
    f32x8 acc[4] = ACC4_INIT;
    gemm22(cat, kw, m0, n0, kb, ks, sl, acc);
#pragma unroll
    for (int mt = 0; mt < 2; ++mt)
#pragma unroll
        for (int nt = 0; nt < 2; ++nt)
#pragma unroll
            for (int r = 0; r < 8; ++r) {
                int mg = m0 + 16 * mt + r + hi, ng = n0 + 16 * nt + sl;
                int t = mg >> 2, b = mg & 3;                 // row = t*B + b
                int nh = ng >> 7, dh = ng & 127;
                kout[(((size_t)(b * NH_ + nh) * TM_ + t) * DH_ + dh)] = f2bf(acc[mt * 2 + nt][r]);
            }
}

// Q projection (last T rows): -> qk = q+k_bias, qr = q+r_bias (bf16, [b][nh][i][dh])
__global__ __launch_bounds__(256) void gemm_q_kernel(
    const uint16_t* __restrict__ catq, const uint16_t* __restrict__ qw,
    const float* __restrict__ kbias, const float* __restrict__ rbias,
    uint16_t* __restrict__ qkout, uint16_t* __restrict__ qrout)
{
    const int lane = threadIdx.x & 31, w = threadIdx.x >> 5;
    const int kb = (lane >> 4) * 8, ks = (lane >> 4) * 16, sl = lane & 15;
    const int hi = (lane >> 4) << 3;
    const int m0 = blockIdx.x * 64 + (w >> 2) * 32;
    const int n0 = blockIdx.y * 128 + (w & 3) * 32;
    f32x8 acc[4] = ACC4_INIT;
    gemm22(catq, qw, m0, n0, kb, ks, sl, acc);
#pragma unroll
    for (int mt = 0; mt < 2; ++mt)
#pragma unroll
        for (int nt = 0; nt < 2; ++nt)
#pragma unroll
            for (int r = 0; r < 8; ++r) {
                int mg = m0 + 16 * mt + r + hi, ng = n0 + 16 * nt + sl;
                int i = mg >> 2, b = mg & 3;
                int nh = ng >> 7, dh = ng & 127;
                size_t o = ((size_t)(b * NH_ + nh) * T_ + i) * DH_ + dh;
                float v = acc[mt * 2 + nt][r];
                qkout[o] = f2bf(v + kbias[ng]);
                qrout[o] = f2bf(v + rbias[ng]);
            }
}

// R projection: pe(2048x1024) @ r_w^T -> rbuf[nh][t][dh] (bf16)
__global__ __launch_bounds__(256) void gemm_r_kernel(
    const uint16_t* __restrict__ pe, const uint16_t* __restrict__ rw,
    uint16_t* __restrict__ rout)
{
    const int lane = threadIdx.x & 31, w = threadIdx.x >> 5;
    const int kb = (lane >> 4) * 8, ks = (lane >> 4) * 16, sl = lane & 15;
    const int hi = (lane >> 4) << 3;
    const int m0 = blockIdx.x * 64 + (w >> 2) * 32;
    const int n0 = blockIdx.y * 128 + (w & 3) * 32;
    f32x8 acc[4] = ACC4_INIT;
    gemm22(pe, rw, m0, n0, kb, ks, sl, acc);
#pragma unroll
    for (int mt = 0; mt < 2; ++mt)
#pragma unroll
        for (int nt = 0; nt < 2; ++nt)
#pragma unroll
            for (int r = 0; r < 8; ++r) {
                int mg = m0 + 16 * mt + r + hi, ng = n0 + 16 * nt + sl;
                int nh = ng >> 7, dh = ng & 127;
                rout[((size_t)nh * TM_ + mg) * DH_ + dh] = f2bf(acc[mt * 2 + nt][r]);
            }
}

// ---------------- fused banded attention + rel_shift + softmax ----------------
// grid: (T/16, B*NH); block 256 (8 waves).
// Dynamic LDS: 16*SROW floats (score band) + SROW ints (cached reset-cumsum).
__global__ __launch_bounds__(256) void attn_kernel(
    const uint16_t* __restrict__ qk, const uint16_t* __restrict__ qr,
    const uint16_t* __restrict__ kbuf, const uint16_t* __restrict__ rbuf,
    const int* __restrict__ cum, float* __restrict__ out)
{
    extern __shared__ char smem[];
    float* S = (float*)smem;                         // [16][SROW]
    int* cumS = (int*)(smem + 16 * SROW * sizeof(float)); // [SROW]
    const int tid = threadIdx.x, lane = tid & 31, w = tid >> 5;
    const int sl = lane & 15, kb = (lane >> 4) * 8, ks = (lane >> 4) * 16;
    const int hi = (lane >> 4) << 3;
    const int i0 = blockIdx.x * 16;
    const int bn = blockIdx.y;                 // b*NH + n
    const int b = bn >> 3, n = bn & 7;

    const uint16_t* Q1 = qk + ((size_t)bn * T_ + i0) * DH_;
    const uint16_t* Q2 = qr + ((size_t)bn * T_ + i0) * DH_;
    const uint16_t* Kp = kbuf + (size_t)bn * TM_ * DH_;
    const uint16_t* Rp = rbuf + (size_t)n * TM_ * DH_;

    // stage the band's reset-cumsum into LDS (read in phase 3; barriers below order it)
    for (int idx = tid; idx < SROW; idx += 256) {
        int j = i0 + idx;
        cumS[idx] = (j < TM_) ? cum[j * B_ + b] : 0x7fffffff;
    }

    // hoist A fragments (query tiles, reused across 130 WMMA tiles)
    bf16x16 aq[4], ar[4];
    {
        const uint16_t* q1row = Q1 + (size_t)sl * DH_;
        const uint16_t* q2row = Q2 + (size_t)sl * DH_;
#pragma unroll
        for (int kk = 0; kk < 4; ++kk) {
            aq[kk] = loadA(q1row, 32 * kk, kb);
            ar[kk] = loadA(q2row, 32 * kk, kb);
        }
    }

    // phase 1: content scores (q+k_bias).k over 66 j-tiles; plain stores cover every column
    for (int lt = w; lt < 66; lt += 8) {
        int j = i0 + 16 * lt + sl;
        if (j > TM_ - 1) j = TM_ - 1;                  // clamped tiles are masked anyway
        const uint16_t* krow = Kp + (size_t)j * DH_;
        f32x8 acc = {0.f,0.f,0.f,0.f,0.f,0.f,0.f,0.f};
#pragma unroll
        for (int kk = 0; kk < 4; ++kk)
            acc = wmma_bf16(aq[kk], loadB(krow, 32 * kk, ks), acc);
#pragma unroll
        for (int r = 0; r < 8; ++r)
            S[(r + hi) * SROW + 16 * lt + sl] = acc[r];
    }
    __syncthreads();

    // phase 2: position scores. rel_shift => r-row index jj = (j-i) + T-1, constant in band coords.
    // S[m][jjloc + m] += P[m][jjloc]; per-tile writes are disjoint across waves/rows.
    for (int jt = w; jt < 64; jt += 8) {
        int jj = (T_ - 1) + 16 * jt + sl;              // 1023 .. 2046
        const uint16_t* rrow = Rp + (size_t)jj * DH_;
        f32x8 acc = {0.f,0.f,0.f,0.f,0.f,0.f,0.f,0.f};
#pragma unroll
        for (int kk = 0; kk < 4; ++kk)
            acc = wmma_bf16(ar[kk], loadB(rrow, 32 * kk, ks), acc);
#pragma unroll
        for (int r = 0; r < 8; ++r) {
            int mrow = r + hi;
            int c = 16 * jt + sl + mrow;               // skew by row: c = jjloc + (i - i0)
            S[mrow * SROW + c] += acc[r];
        }
    }
    __syncthreads();

    // phase 3: masked softmax per row; each wave handles 2 rows; full 2048-wide output
    const float scl = 0.08838834764831845f;            // 1/sqrt(128)
    for (int rr = 0; rr < 2; ++rr) {
        int ri = 2 * w + rr;
        int i = i0 + ri;
        int cq = cumS[ri + 1024];                      // cum[(MEM+i)*B+b]: j=i+1024 is in-band
        float mx = -3.0e38f;
        for (int c = lane; c < SROW; c += 32) {
            bool valid = (c > ri) && (c < ri + 1024) && (cumS[c] >= cq);
            if (valid) {
                float s = fminf(fmaxf(S[ri * SROW + c] * scl, -65000.f), 65000.f);
                mx = fmaxf(mx, s);
            }
        }
#pragma unroll
        for (int off = 16; off; off >>= 1) mx = fmaxf(mx, __shfl_xor(mx, off, 32));
        float sum = 0.f;
        for (int c = lane; c < SROW; c += 32) {
            bool valid = (c > ri) && (c < ri + 1024) && (cumS[c] >= cq);
            if (valid) {
                float s = fminf(fmaxf(S[ri * SROW + c] * scl, -65000.f), 65000.f);
                float e = __expf(s - mx);
                S[ri * SROW + c] = e;                  // row is wave-private: cache exp in place
                sum += e;
            }
        }
#pragma unroll
        for (int off = 16; off; off >>= 1) sum += __shfl_xor(sum, off, 32);
        bool any = (mx > -1.0e38f);
        float inv = (any && sum > 0.f) ? (1.0f / sum) : 0.0f;
        for (int j = lane; j < TM_; j += 32) {
            float p = 0.f;
            int c = j - i0;
            if (any && c > ri && c < ri + 1024 && cumS[c] >= cq)
                p = S[ri * SROW + c] * inv;
            out[(((size_t)i * TM_ + j) * B_ + b) * NH_ + n] = p;
        }
    }
}

// ---------------- host launcher ----------------
extern "C" void kernel_launch(void* const* d_in, const int* in_sizes, int n_in,
                              void* d_out, int out_size, void* d_ws, size_t ws_size,
                              hipStream_t stream) {
    const float* h        = (const float*)d_in[0];
    const float* m        = (const float*)d_in[1];
    const float* q_w      = (const float*)d_in[2];
    const float* k_w      = (const float*)d_in[3];
    const float* r_w      = (const float*)d_in[4];
    const float* k_bias   = (const float*)d_in[5];
    const float* r_bias   = (const float*)d_in[6];
    const float* ln_scale = (const float*)d_in[7];
    const float* ln_bias  = (const float*)d_in[8];
    const int*   reset    = (const int*)d_in[9];
    const int*   oreset   = (const int*)d_in[10];
    float* out = (float*)d_out;

    // carve workspace (all 16B-aligned)
    char* w = (char*)d_ws;
    uint16_t* catb = (uint16_t*)w;  w += (size_t)TM_ * B_ * D_ * 2;      // 16 MB
    uint16_t* peb  = (uint16_t*)w;  w += (size_t)TM_ * D_ * 2;           // 4 MB
    uint16_t* qwb  = (uint16_t*)w;  w += (size_t)D_ * D_ * 2;            // 2 MB
    uint16_t* kwb  = (uint16_t*)w;  w += (size_t)D_ * D_ * 2;            // 2 MB
    uint16_t* rwb  = (uint16_t*)w;  w += (size_t)D_ * D_ * 2;            // 2 MB
    uint16_t* qkb  = (uint16_t*)w;  w += (size_t)B_ * NH_ * T_ * DH_ * 2;  // 8 MB
    uint16_t* qrb  = (uint16_t*)w;  w += (size_t)B_ * NH_ * T_ * DH_ * 2;  // 8 MB
    uint16_t* kbuf = (uint16_t*)w;  w += (size_t)B_ * NH_ * TM_ * DH_ * 2; // 16 MB
    uint16_t* rbuf = (uint16_t*)w;  w += (size_t)NH_ * TM_ * DH_ * 2;      // 4 MB
    int*      cum  = (int*)w;       w += (size_t)TM_ * B_ * 4;

    // stage 0: layernorm + PE + weight cvt + cumsum
    ln_kernel<<<TM_ * B_, 256, 0, stream>>>(h, m, ln_scale, ln_bias, catb);
    pe_kernel<<<TM_, 256, 0, stream>>>(peb);
    cvt_kernel<<<(D_ * D_ + 255) / 256, 256, 0, stream>>>(q_w, qwb, D_ * D_);
    cvt_kernel<<<(D_ * D_ + 255) / 256, 256, 0, stream>>>(k_w, kwb, D_ * D_);
    cvt_kernel<<<(D_ * D_ + 255) / 256, 256, 0, stream>>>(r_w, rwb, D_ * D_);
    cum_kernel<<<1, 32, 0, stream>>>(reset, oreset, cum);

    // stage 1: projections (bf16 WMMA, f32 accumulate, 2x2 tiles per wave)
    gemm_k_kernel<<<dim3((TM_ * B_) / 64, D_ / 128), 256, 0, stream>>>(catb, kwb, kbuf);
    gemm_q_kernel<<<dim3((T_ * B_) / 64, D_ / 128), 256, 0, stream>>>(
        catb + (size_t)MEM_ * B_ * D_, qwb, k_bias, r_bias, qkb, qrb);
    gemm_r_kernel<<<dim3(TM_ / 64, D_ / 128), 256, 0, stream>>>(peb, rwb, rbuf);

    // stage 2: fused banded attention + rel_shift + masked softmax
    size_t attn_lds = (size_t)16 * SROW * sizeof(float) + SROW * sizeof(int);
    attn_kernel<<<dim3(T_ / 16, B_ * NH_), 256, attn_lds, stream>>>(
        qkb, qrb, kbuf, rbuf, cum, out);
}
